// DecoderHead_50714973831863
// MI455X (gfx1250) — matrix operand order repaired and checked by reading, not compiled, not run on hardware
//
#include <hip/hip_runtime.h>

typedef __attribute__((ext_vector_type(16))) _Float16 v16h;
typedef __attribute__((ext_vector_type(8)))  _Float16 v8h;
typedef __attribute__((ext_vector_type(8)))  float    v8f;

#define WMMA_F32_F16(a, b, c) \
  __builtin_amdgcn_wmma_f32_16x16x32_f16(false, (a), false, (b), (short)0, (c), false, false)

static __device__ __forceinline__ v8f vzero8() {
  v8f z;
  for (int i = 0; i < 8; ++i) z[i] = 0.0f;
  return z;
}

// ---------------------------------------------------------------------------
// Kernel 1: fused QKV projection. y = x @ W, fp32 -> f16 outputs.
//   x  : [B*T, C] fp32         (B*T = 16384, C = 1024)
//   W  : [C, H]   fp32         (H = 128), selected by blockIdx.y (0:Q 1:K 2:V)
//   q16/k16 : [B*T, H] f16 row-major
//   vt      : [B][H][T] f16 H-major (so attention B-frags over keys are contiguous)
// Block = 256 threads (8 waves); each wave computes a 16x128 output tile.
// ---------------------------------------------------------------------------
__global__ __launch_bounds__(256) void qkv_proj_kernel(
    const float* __restrict__ x,
    const float* __restrict__ Wq, const float* __restrict__ Wk, const float* __restrict__ Wv,
    _Float16* __restrict__ q16, _Float16* __restrict__ k16, _Float16* __restrict__ vt)
{
  constexpr int Cdim = 1024, Hdim = 128, Tdim = 4096;
  const int proj = blockIdx.y;
  const float* __restrict__ W = (proj == 0) ? Wq : (proj == 1) ? Wk : Wv;

  __shared__ __align__(32) _Float16 Wt[128][32];   // W tile transposed: [n][k]

  const int tid  = threadIdx.x;
  const int wave = tid >> 5;
  const int lane = tid & 31;
  const int hi   = lane >> 4;      // half-wave select
  const int ln   = lane & 15;

  const int m0  = blockIdx.x * 128 + wave * 16;    // wave's first output row
  const int row = m0 + ln;                         // A-layout: M = lane & 15

  v8f acc[8];
  for (int nt = 0; nt < 8; ++nt) acc[nt] = vzero8();

  for (int kk = 0; kk < Cdim; kk += 32) {
    // Stage W[kk..kk+32)[0..128) transposed into LDS as f16 (coalesced reads).
    for (int j = 0; j < 16; ++j) {
      int e = tid + j * 256;       // 0..4095
      int k = e >> 7;              // 0..31
      int n = e & 127;
      Wt[n][k] = (_Float16)W[(size_t)(kk + k) * Hdim + n];
    }
    __syncthreads();

    // A fragment 16x32 f16, per the ISA layout:
    // lanes 0-15: K = {kk+0..7, kk+16..23}; lanes 16-31: K = {kk+8..15, kk+24..31}
    v16h a;
    {
      const float* xp = x + (size_t)row * Cdim + kk + hi * 8;
      float4 f0 = *(const float4*)(xp + 0);
      float4 f1 = *(const float4*)(xp + 4);
      float4 f2 = *(const float4*)(xp + 16);
      float4 f3 = *(const float4*)(xp + 20);
      a[0]  = (_Float16)f0.x; a[1]  = (_Float16)f0.y; a[2]  = (_Float16)f0.z; a[3]  = (_Float16)f0.w;
      a[4]  = (_Float16)f1.x; a[5]  = (_Float16)f1.y; a[6]  = (_Float16)f1.z; a[7]  = (_Float16)f1.w;
      a[8]  = (_Float16)f2.x; a[9]  = (_Float16)f2.y; a[10] = (_Float16)f2.z; a[11] = (_Float16)f2.w;
      a[12] = (_Float16)f3.x; a[13] = (_Float16)f3.y; a[14] = (_Float16)f3.z; a[15] = (_Float16)f3.w;
    }

    // B fragments: lane -> column N, K order matches A: {hi*8..+7, 16+hi*8..+7} in Wt row.
    for (int nt = 0; nt < 8; ++nt) {
      v8h blo = *(const v8h*)&Wt[nt * 16 + ln][hi * 8];
      v8h bhi = *(const v8h*)&Wt[nt * 16 + ln][hi * 8 + 16];
      v16h b = __builtin_shufflevector(blo, bhi, 0,1,2,3,4,5,6,7,8,9,10,11,12,13,14,15);
      acc[nt] = WMMA_F32_F16(a, b, acc[nt]);
    }
    __syncthreads();
  }

  // Epilogue: uniform branch hoisted OUTSIDE the store loops (proj is uniform).
  // C/D layout: VGPR r holds row (r + hi*8), lane&15 is column.
  if (proj == 2) {
    // V stored H-major per batch: vt[(b*H + col)*T + t]
    for (int nt = 0; nt < 8; ++nt) {
      int col = nt * 16 + ln;
      for (int r = 0; r < 8; ++r) {
        int g = m0 + r + hi * 8;
        int b = g >> 12;              // g / T
        int t = g & (Tdim - 1);       // g % T
        vt[((size_t)(b * Hdim + col)) * Tdim + t] = (_Float16)acc[nt][r];
      }
    }
  } else {
    _Float16* __restrict__ O = (proj == 0) ? q16 : k16;
    for (int nt = 0; nt < 8; ++nt) {
      int col = nt * 16 + ln;
      for (int r = 0; r < 8; ++r) {
        int g = m0 + r + hi * 8;
        O[(size_t)g * Hdim + col] = (_Float16)acc[nt][r];
      }
    }
  }
}

// ---------------------------------------------------------------------------
// Kernel 2: causal flash attention. Each wave owns a 16-query tile; 8 waves
// per block => 128 queries/block; grid = B * T/128 = 128 blocks.
// ---------------------------------------------------------------------------
__global__ __launch_bounds__(256) void flash_attn_kernel(
    const _Float16* __restrict__ q16, const _Float16* __restrict__ k16,
    const _Float16* __restrict__ vt, float* __restrict__ out)
{
  constexpr int Tdim = 4096, Hdim = 128;
  const int tid  = threadIdx.x;
  const int wave = tid >> 5;
  const int lane = tid & 31;
  const int hi   = lane >> 4;
  const int ln   = lane & 15;

  const int qtilesPerBatch = Tdim / 128;               // 32
  const int b  = blockIdx.x / qtilesPerBatch;
  const int q0 = (blockIdx.x % qtilesPerBatch) * 128 + wave * 16;  // within batch

  const _Float16* __restrict__ Qb = q16 + (size_t)b * Tdim * Hdim;
  const _Float16* __restrict__ Kb = k16 + (size_t)b * Tdim * Hdim;
  const _Float16* __restrict__ Vb = vt  + (size_t)b * Hdim * Tdim;

  __shared__ __align__(32) _Float16 Plds[8][16][32];   // per-wave P staging

  // Q fragments: 4 K-chunks of 32 over H=128 (A-layout, loaded once).
  v16h qf[4];
  for (int c = 0; c < 4; ++c) {
    const _Float16* qp = Qb + (size_t)(q0 + ln) * Hdim + c * 32 + hi * 8;
    v8h lo = *(const v8h*)(qp);
    v8h hh = *(const v8h*)(qp + 16);
    qf[c] = __builtin_shufflevector(lo, hh, 0,1,2,3,4,5,6,7,8,9,10,11,12,13,14,15);
  }

  v8f o[8];
  for (int nt = 0; nt < 8; ++nt) o[nt] = vzero8();
  float Mrow[8], Lrow[8];
  for (int r = 0; r < 8; ++r) { Mrow[r] = -1e30f; Lrow[r] = 0.0f; }

  const float scale = 0.088388347648318447f;           // H^-0.5 = 1/sqrt(128)
  const int smax = q0 + 15;                            // causal upper bound

  for (int s0 = 0; s0 <= smax; s0 += 32) {
    // ---- S = Q * K^T for two adjacent 16-key tiles (keys s0.. and s0+16..)
    v8f S0 = vzero8(), S1 = vzero8();
    for (int c = 0; c < 4; ++c) {
      const _Float16* kp = Kb + (size_t)(s0 + ln) * Hdim + c * 32 + hi * 8;
      v16h b0 = __builtin_shufflevector(*(const v8h*)(kp), *(const v8h*)(kp + 16), 0,1,2,3,4,5,6,7,8,9,10,11,12,13,14,15);
      v16h b1 = __builtin_shufflevector(*(const v8h*)(kp + (size_t)16 * Hdim), *(const v8h*)(kp + (size_t)16 * Hdim + 16), 0,1,2,3,4,5,6,7,8,9,10,11,12,13,14,15);
      S0 = WMMA_F32_F16(qf[c], b0, S0);
      S1 = WMMA_F32_F16(qf[c], b1, S1);
    }

    // ---- scale + causal mask + online softmax (row == one VGPR across a half-wave)
    float alpha[8];
    for (int r = 0; r < 8; ++r) {
      int qrow = q0 + r + hi * 8;
      float a0 = S0[r] * scale;
      float a1 = S1[r] * scale;
      if (s0 + ln      > qrow) a0 = -1e30f;
      if (s0 + 16 + ln > qrow) a1 = -1e30f;
      float m = fmaxf(a0, a1);
      for (int off = 8; off >= 1; off >>= 1) m = fmaxf(m, __shfl_xor(m, off, 16));
      float mnew = fmaxf(Mrow[r], m);
      float p0 = __expf(a0 - mnew);
      float p1 = __expf(a1 - mnew);
      S0[r] = p0; S1[r] = p1;
      float rs = p0 + p1;
      for (int off = 8; off >= 1; off >>= 1) rs += __shfl_xor(rs, off, 16);
      alpha[r] = __expf(Mrow[r] - mnew);
      Lrow[r]  = Lrow[r] * alpha[r] + rs;
      Mrow[r]  = mnew;
    }
    for (int nt = 0; nt < 8; ++nt)
      for (int r = 0; r < 8; ++r) o[nt][r] *= alpha[r];

    // ---- P (C-layout f32) -> A-layout f16 fragment via per-wave LDS round trip
    for (int r = 0; r < 8; ++r) {
      Plds[wave][r + hi * 8][ln]      = (_Float16)S0[r];
      Plds[wave][r + hi * 8][16 + ln] = (_Float16)S1[r];
    }
    v16h pf;
    {
      v8h lo = *(const v8h*)&Plds[wave][ln][hi * 8];
      v8h hh = *(const v8h*)&Plds[wave][ln][hi * 8 + 16];
      pf = __builtin_shufflevector(lo, hh, 0,1,2,3,4,5,6,7,8,9,10,11,12,13,14,15);
    }

    // ---- O += P * V ; V is H-major so B-frags are two 16B loads of contiguous keys
    for (int nt = 0; nt < 8; ++nt) {
      const _Float16* vp = Vb + (size_t)(nt * 16 + ln) * Tdim + s0 + hi * 8;
      v16h vf = __builtin_shufflevector(*(const v8h*)(vp), *(const v8h*)(vp + 16), 0,1,2,3,4,5,6,7,8,9,10,11,12,13,14,15);
      o[nt] = WMMA_F32_F16(pf, vf, o[nt]);
    }
  }

  // ---- normalize and store fp32 output [B,T,H]
  for (int nt = 0; nt < 8; ++nt) {
    for (int r = 0; r < 8; ++r) {
      int m = q0 + r + hi * 8;
      out[((size_t)b * Tdim + m) * Hdim + nt * 16 + ln] = o[nt][r] / Lrow[r];
    }
  }
}

// ---------------------------------------------------------------------------
extern "C" void kernel_launch(void* const* d_in, const int* in_sizes, int n_in,
                              void* d_out, int out_size, void* d_ws, size_t ws_size,
                              hipStream_t stream) {
  (void)in_sizes; (void)n_in; (void)out_size; (void)ws_size;
  // setup_inputs order: x, Wk, Wq, Wv  (all fp32)
  const float* x  = (const float*)d_in[0];
  const float* Wk = (const float*)d_in[1];
  const float* Wq = (const float*)d_in[2];
  const float* Wv = (const float*)d_in[3];
  float* out = (float*)d_out;

  constexpr size_t BT = 4 * 4096;   // 16384 rows
  constexpr size_t H  = 128;
  _Float16* q16 = (_Float16*)d_ws;             // [BT, H]
  _Float16* k16 = q16 + BT * H;                // [BT, H]
  _Float16* vt  = k16 + BT * H;                // [B][H][T]

  // 128 row-tiles x 3 projections
  qkv_proj_kernel<<<dim3(128, 3), 256, 0, stream>>>(x, Wq, Wk, Wv, q16, k16, vt);
  // B * T/128 = 128 blocks, 8 waves each
  flash_attn_kernel<<<dim3(128), 256, 0, stream>>>(q16, k16, vt, out);
}
